// UnitWiseGRU_4715874091019
// MI455X (gfx1250) — matrix operand order repaired and checked by reading, not compiled, not run on hardware
//
#include <hip/hip_runtime.h>
#include <hip/hip_bf16.h>

typedef __bf16 bf16;
typedef __attribute__((ext_vector_type(16))) __bf16 v16bf;
typedef __attribute__((ext_vector_type(8)))  __bf16 v8bf;
typedef __attribute__((ext_vector_type(2)))  __bf16 v2bf;
typedef __attribute__((ext_vector_type(8)))  float  v8f;
typedef __attribute__((ext_vector_type(4)))  float  v4f;
typedef __attribute__((ext_vector_type(2)))  float  v2f;

#define NU     64    // units
#define NI     256   // in features  (K total)
#define NO     256   // out features
#define NB     256   // batch (M total)
#define KSTEP  32    // bf16 WMMA K
#define NTILE  64    // N per workgroup
#define LDK    40    // padded halves per LDS row (80B rows: 16B aligned, conflict-free A loads)

__global__ __launch_bounds__(512)
void gru_unit_wmma_kernel(const float* __restrict__ x,
                          const float* __restrict__ w_ir, const float* __restrict__ w_iz,
                          const float* __restrict__ w_in,
                          const float* __restrict__ w_hr, const float* __restrict__ w_hz,
                          const float* __restrict__ w_hn,
                          const float* __restrict__ h0,
                          float* __restrict__ out)
{
    __shared__ __align__(16) bf16 lds_x[NB * LDK];        // 20480 B : X tile  [m][k] bf16
    __shared__ __align__(16) bf16 lds_w[3][NTILE * LDK];  // 15360 B : W tiles [n][k] bf16 (transposed)
    __shared__ float lds_hv[3 * NTILE];                   // h0 @ W_h  per gate, this o-tile
    __shared__ float lds_h0[NTILE];                       // h0 values for this o-tile

    const int t     = threadIdx.x;        // 0..511
    const int u     = blockIdx.y;         // unit
    const int o0    = blockIdx.x * NTILE; // N tile origin
    const int lane  = t & 31;
    const int wv    = t >> 5;             // wave 0..15
    const int m0    = wv * 16;            // this wave's M strip
    const int lhalf = lane >> 4;          // 0/1
    const int lrow  = lane & 15;

    const float* wi[3] = { w_ir, w_iz, w_in };
    const float* wh[3] = { w_hr, w_hz, w_hn };

    // ---- tiny h-side GEMV: hv[g][n] = sum_i h0[u][i] * w_h[g][u][i][o0+n]  (h broadcast over batch)
    if (t < 3 * NTILE) {
        const int g = t >> 6, n = t & 63;
        const float* w = wh[g] + (size_t)u * NI * NO + (size_t)(o0 + n);
        const float* h = h0 + (size_t)u * NO;
        float acc = 0.f;
        #pragma unroll 4
        for (int i = 0; i < NO; ++i)
            acc = fmaf(h[i], w[(size_t)i * NO], acc);
        lds_hv[t] = acc;
    }
    if (t < NTILE) lds_h0[t] = h0[(size_t)u * NO + o0 + t];

    const v8f vzero = { 0.f, 0.f, 0.f, 0.f, 0.f, 0.f, 0.f, 0.f };
    v8f acc[3][4];
    #pragma unroll
    for (int g = 0; g < 3; ++g)
        #pragma unroll
        for (int ns = 0; ns < 4; ++ns)
            acc[g][ns] = vzero;

    // W-staging 2x2 micro-tile coords: o2 = even o, k2 = even k
    const int o2 = (t & 31) * 2;          // 0..62 (per-wave contiguous in o)
    const int k2 = (t >> 5) * 2;          // 0..30

    // ---- main K loop over the x-side GEMMs (3 gates fused)
    for (int k0 = 0; k0 < NI; k0 += KSTEP) {
        __syncthreads();   // fragments of previous step consumed

        // stage X tile [256 x 32] f32 -> bf16 LDS (row-major, padded); packed converts + b64 stores
        #pragma unroll
        for (int it = 0; it < 4; ++it) {
            const int f  = t + it * 512;       // 0..2047 quads
            const int b  = f >> 3;
            const int kq = f & 7;
            v4f v = *(const v4f*)(x + ((size_t)b * NU + u) * NI + k0 + kq * 4);
            bf16* d = &lds_x[b * LDK + kq * 4];
            d[0] = (bf16)v.x; d[1] = (bf16)v.y; d[2] = (bf16)v.z; d[3] = (bf16)v.w;
        }

        // stage W tiles [32 x 64] f32 -> transposed bf16 LDS [n][k] in 2x2 micro-tiles:
        // two v2f loads (rows k2,k2+1), two packed v2bf stores (rows o2,o2+1), pairs along k.
        #pragma unroll
        for (int g = 0; g < 3; ++g) {
            const float* wb = wi[g] + (size_t)u * NI * NO + o0;
            v2f a0 = *(const v2f*)(wb + (size_t)(k0 + k2)     * NO + o2);
            v2f a1 = *(const v2f*)(wb + (size_t)(k0 + k2 + 1) * NO + o2);
            v2bf p0, p1;
            p0.x = (bf16)a0.x; p0.y = (bf16)a1.x;   // row o2:   (k2, k2+1)
            p1.x = (bf16)a0.y; p1.y = (bf16)a1.y;   // row o2+1: (k2, k2+1)
            *(v2bf*)&lds_w[g][(o2    ) * LDK + k2] = p0;
            *(v2bf*)&lds_w[g][(o2 + 1) * LDK + k2] = p1;
        }
        __syncthreads();

        // A fragment (ISA 16-bit A layout): lane holds M=m0+lrow, k = (e<8?e:e+8) + 8*lhalf
        const int arow = m0 + lrow;
        v8bf alo = *(const v8bf*)&lds_x[arow * LDK + lhalf * 8];
        v8bf ahi = *(const v8bf*)&lds_x[arow * LDK + 16 + lhalf * 8];
        v16bf afrag = __builtin_shufflevector(alo, ahi,
                        0,1,2,3,4,5,6,7,8,9,10,11,12,13,14,15);

        #pragma unroll
        for (int g = 0; g < 3; ++g) {
            #pragma unroll
            for (int ns = 0; ns < 4; ++ns) {
                // B fragment (ISA B layout): lane holds N=ns*16+lrow, k = (e<8?e:e+8) + 8*lhalf
                const int nrow = ns * 16 + lrow;
                const bf16* bp = &lds_w[g][nrow * LDK + lhalf * 8];
                v8bf blo = *(const v8bf*)(bp);
                v8bf bhi = *(const v8bf*)(bp + 16);
                v16bf bfrag = __builtin_shufflevector(blo, bhi,
                                0,1,2,3,4,5,6,7,8,9,10,11,12,13,14,15);
                acc[g][ns] = __builtin_amdgcn_wmma_f32_16x16x32_bf16(
                    false, afrag, false, bfrag, (short)0, acc[g][ns], false, false);
            }
        }
    }

    // ---- GRU gating epilogue (C/D layout: vgpr v -> m = m0 + v + 8*lhalf, n = lrow)
    const size_t outUO = (size_t)NU * NO;
    #pragma unroll
    for (int ns = 0; ns < 4; ++ns) {
        const int nl   = ns * 16 + lrow;
        const float hr = lds_hv[nl];
        const float hz = lds_hv[64 + nl];
        const float hn = lds_hv[128 + nl];
        const float hp = lds_h0[nl];
        #pragma unroll
        for (int v = 0; v < 8; ++v) {
            const int b = m0 + v + lhalf * 8;
            const float xr = acc[0][ns][v] + hr;
            const float xz = acc[1][ns][v] + hz;
            const float xn = acc[2][ns][v];
            const float r  = 1.f / (1.f + __expf(-xr));
            const float z  = 1.f / (1.f + __expf(-xz));
            const float e2 = __expf(2.f * (xn + r * hn));
            const float nn = 1.f - 2.f / (e2 + 1.f);       // tanh, inf-safe
            out[(size_t)b * outUO + (size_t)u * NO + o0 + nl] = (1.f - z) * nn + z * hp;
        }
    }
}

extern "C" void kernel_launch(void* const* d_in, const int* in_sizes, int n_in,
                              void* d_out, int out_size, void* d_ws, size_t ws_size,
                              hipStream_t stream) {
    (void)in_sizes; (void)n_in; (void)out_size; (void)d_ws; (void)ws_size;
    dim3 grid(NO / NTILE, NU);   // (4, 64) = 256 workgroups
    dim3 block(512);             // 16 waves (wave32)
    gru_unit_wmma_kernel<<<grid, block, 0, stream>>>(
        (const float*)d_in[0],
        (const float*)d_in[1], (const float*)d_in[2], (const float*)d_in[3],
        (const float*)d_in[4], (const float*)d_in[5], (const float*)d_in[6],
        (const float*)d_in[7],
        (float*)d_out);
}
